// MultiheadAttention_58926951301232
// MI455X (gfx1250) — matrix operand order repaired and checked by reading, not compiled, not run on hardware
//
#include <hip/hip_runtime.h>

// MI455X (gfx1250) multi-head attention forward.
// Pipeline: (1) one-shot f32->bf16 conversion of activations+weights,
// (2) unified bf16 GEMM (v_wmma_f32_16x16x32_bf16, double-buffered LDS,
// pure b128 copy staging), (3) two-pass attention with per-lane online
// softmax stats (no per-tile cross-lane reductions), (4) bf16 GEMM with
// f32 output for the final projection.

#define D_MODEL 1024
#define NB      8
#define SEQ     1024
#define NH      16
#define DK      64
#define LSTR    40   // LDS row stride in bf16 elements (80B, 16B-aligned)

typedef __attribute__((ext_vector_type(16))) __bf16 v16bf;
typedef __attribute__((ext_vector_type(8)))  float  v8f;

union Frag16 { v16bf v; uint4 q[2]; unsigned short u[16]; };

__device__ __forceinline__ unsigned short f32_to_bf16(float f) {
  __bf16 h = (__bf16)f;                       // native v_cvt_pk_bf16_f32 on gfx1250
  return __builtin_bit_cast(unsigned short, h);
}

__device__ __forceinline__ v8f v8f_zero() {
  v8f z;
#pragma unroll
  for (int i = 0; i < 8; ++i) z[i] = 0.0f;
  return z;
}

__device__ __forceinline__ v8f wmma_bf16(v16bf a, v16bf b, v8f c) {
  // (neg_a, A, neg_b, B, c_mod, C, reuse_a, reuse_b)
  return __builtin_amdgcn_wmma_f32_16x16x32_bf16(false, a, false, b, (short)0, c,
                                                 false, false);
}

// ---------------------------------------------------------------------------
// One-shot f32 -> bf16 conversion (memory bound; feeds all GEMMs).
// n4 = element count / 4.
// ---------------------------------------------------------------------------
__global__ __launch_bounds__(256)
void cvt_f32_bf16(const float* __restrict__ src, unsigned short* __restrict__ dst,
                  int n4) {
  int idx = blockIdx.x * blockDim.x + threadIdx.x;
  const int stride = gridDim.x * blockDim.x;
  for (; idx < n4; idx += stride) {
    float4 f = reinterpret_cast<const float4*>(src)[idx];
    unsigned int lo = (unsigned int)f32_to_bf16(f.x) |
                      ((unsigned int)f32_to_bf16(f.y) << 16);
    unsigned int hi = (unsigned int)f32_to_bf16(f.z) |
                      ((unsigned int)f32_to_bf16(f.w) << 16);
    uint2 o; o.x = lo; o.y = hi;
    reinterpret_cast<uint2*>(dst)[idx] = o;
  }
}

// ---------------------------------------------------------------------------
// C[m,n] = sum_k A[m,k]*W[n,k] + bias[n]; A bf16 [8192,1024], W bf16 [1024,1024].
// OUT_F32 ? f32 output : bf16 output. Grid (8192/64, 1024/64), block 128.
// Double-buffered LDS; staging is a pure b128 copy (no conversions in loop).
// ---------------------------------------------------------------------------
template <bool OUT_F32>
__global__ __launch_bounds__(128)
void gemm_bf16(const unsigned short* __restrict__ A,
               const unsigned short* __restrict__ W,
               const float* __restrict__ bias, void* __restrict__ Cv) {
  __shared__ __align__(16) unsigned short lA[2][64 * LSTR];
  __shared__ __align__(16) unsigned short lB[2][64 * LSTR];
  const int t = threadIdx.x;
  const int wv = t >> 5, lane = t & 31;
  const int laneLo = lane & 15, laneHi = lane >> 4;
  const int m0 = blockIdx.x * 64, n0 = blockIdx.y * 64;
  const int row = t >> 1, half = (t & 1) * 16;

  const unsigned short* srcA0 = A + (size_t)(m0 + row) * D_MODEL + half;
  const unsigned short* srcB0 = W + (size_t)(n0 + row) * D_MODEL + half;

  v8f acc[4];
#pragma unroll
  for (int nt = 0; nt < 4; ++nt) acc[nt] = v8f_zero();

  uint4 ua[2], ub[2];
  auto fetch = [&](int k0) {
    ua[0] = reinterpret_cast<const uint4*>(srcA0 + k0)[0];
    ua[1] = reinterpret_cast<const uint4*>(srcA0 + k0)[1];
    ub[0] = reinterpret_cast<const uint4*>(srcB0 + k0)[0];
    ub[1] = reinterpret_cast<const uint4*>(srcB0 + k0)[1];
  };
  auto commit = [&](int buf) {
    uint4* dstA = reinterpret_cast<uint4*>(lA[buf] + row * LSTR + half);
    uint4* dstB = reinterpret_cast<uint4*>(lB[buf] + row * LSTR + half);
    dstA[0] = ua[0]; dstA[1] = ua[1];
    dstB[0] = ub[0]; dstB[1] = ub[1];
  };
  auto compute = [&](int buf) {
    // A fragment 16x32: lane M=laneLo, element e -> K = laneHi*16 + e
    Frag16 a;
    const uint4* ap = reinterpret_cast<const uint4*>(
        lA[buf] + (wv * 16 + laneLo) * LSTR + laneHi * 16);
    a.q[0] = ap[0];
    a.q[1] = ap[1];
#pragma unroll
    for (int nt = 0; nt < 4; ++nt) {
      // B fragment 32x16: lane N=laneLo, element e -> K = laneHi*16 + e
      Frag16 bfr;
      const uint4* bp = reinterpret_cast<const uint4*>(
          lB[buf] + (nt * 16 + laneLo) * LSTR + laneHi * 16);
      bfr.q[0] = bp[0];
      bfr.q[1] = bp[1];
      acc[nt] = wmma_bf16(a.v, bfr.v, acc[nt]);
    }
  };

  fetch(0);
  commit(0);
  __syncthreads();
  int cur = 0;
  for (int k0 = 32; k0 < D_MODEL; k0 += 32) {
    fetch(k0);          // global loads for next panel (in flight during WMMA)
    compute(cur);       // WMMAs on current panel
    commit(cur ^ 1);    // stage next panel
    __syncthreads();
    cur ^= 1;
  }
  compute(cur);

#pragma unroll
  for (int nt = 0; nt < 4; ++nt) {
    const int col = n0 + nt * 16 + laneLo;
    const float bv = bias[col];
#pragma unroll
    for (int i = 0; i < 8; ++i) {   // C/D layout: VGPR i -> rows i / 8+i
      const int r = m0 + wv * 16 + laneHi * 8 + i;
      if (OUT_F32)
        reinterpret_cast<float*>(Cv)[(size_t)r * D_MODEL + col] = acc[nt][i] + bv;
      else
        reinterpret_cast<unsigned short*>(Cv)[(size_t)r * D_MODEL + col] =
            f32_to_bf16(acc[nt][i] + bv);
    }
  }
}

// ---------------------------------------------------------------------------
// Attention core. Grid (S/64, H, B), block 128 (4 waves; wave = 16 q-rows).
// Pass 1: per-lane online (max, sum) over each lane's own column slice;
// single cross-lane merge at the end. Pass 2: recompute logits, emit f32
// scores, accumulate P @ V via WMMA.
// ---------------------------------------------------------------------------
__global__ __launch_bounds__(128)
void attention(const unsigned short* __restrict__ qb,
               const unsigned short* __restrict__ kb,
               const unsigned short* __restrict__ vb,
               float* __restrict__ scores,
               unsigned short* __restrict__ attnb) {
  const int qblk = blockIdx.x;         // 0..15
  const int h    = blockIdx.y;         // 0..15
  const int bb   = blockIdx.z;         // 0..7
  const int t = threadIdx.x;
  const int wv = t >> 5, lane = t & 31;
  const int laneLo = lane & 15, laneHi = lane >> 4;
  const size_t baseRow = (size_t)bb * SEQ;
  const int q0 = qblk * 64 + wv * 16;
  const int dh = h * DK;

  __shared__ __align__(16) unsigned short vt[64 * LSTR];      // V^T tile [d][key 0..31]
  __shared__ __align__(16) unsigned short pt[4][16 * LSTR];   // per-wave P tile [m][k 0..31]

  // Q fragments for both 32-wide d-chunks (kept in registers for both passes)
  Frag16 aq[2];
  {
    const unsigned short* qrow = qb + (baseRow + q0 + laneLo) * D_MODEL + dh;
#pragma unroll
    for (int c = 0; c < 2; ++c) {
      const uint4* p = reinterpret_cast<const uint4*>(qrow + c * 32 + laneHi * 16);
      aq[c].q[0] = p[0];
      aq[c].q[1] = p[1];
    }
  }

  // ---- pass 1: per-lane online max / sum(exp); no cross-lane traffic ----
  float pm[8], ps[8];
#pragma unroll
  for (int i = 0; i < 8; ++i) { pm[i] = -1e30f; ps[i] = 0.0f; }

  for (int kt = 0; kt < 64; ++kt) {
    const int n0 = kt * 16;
    Frag16 bk[2];
    const unsigned short* krow = kb + (baseRow + n0 + laneLo) * D_MODEL + dh;
    if (kt < 63)   // prefetch next key tile row (global_prefetch_b8)
      __builtin_prefetch(krow + 16 * D_MODEL, 0, 1);
#pragma unroll
    for (int c = 0; c < 2; ++c) {
      const uint4* p = reinterpret_cast<const uint4*>(krow + c * 32 + laneHi * 16);
      bk[c].q[0] = p[0];
      bk[c].q[1] = p[1];
    }
    v8f l = v8f_zero();
    l = wmma_bf16(aq[0].v, bk[0].v, l);
    l = wmma_bf16(aq[1].v, bk[1].v, l);
#pragma unroll
    for (int i = 0; i < 8; ++i) {
      const float x = l[i] * 0.125f;                  // 1/sqrt(64)
      const float nm = fmaxf(pm[i], x);
      ps[i] = ps[i] * __expf(pm[i] - nm) + __expf(x - nm);
      pm[i] = nm;
    }
  }
  // merge the 16 per-lane partials of each row (once per wave)
  float rmax[8], rinv[8];
#pragma unroll
  for (int i = 0; i < 8; ++i) {
#pragma unroll
    for (int mask = 1; mask < 16; mask <<= 1) {
      const float om = __shfl_xor(pm[i], mask, 32);
      const float os = __shfl_xor(ps[i], mask, 32);
      const float nm = fmaxf(pm[i], om);
      ps[i] = ps[i] * __expf(pm[i] - nm) + os * __expf(om - nm);
      pm[i] = nm;
    }
    rmax[i] = pm[i];
    rinv[i] = 1.0f / ps[i];
  }

  // ---- pass 2: recompute logits, emit scores, accumulate P @ V ----
  v8f oacc[4];
#pragma unroll
  for (int nt = 0; nt < 4; ++nt) oacc[nt] = v8f_zero();
  unsigned short* myPt = pt[wv];

  for (int kt2 = 0; kt2 < 32; ++kt2) {
    const int k0 = kt2 * 32;
    __syncthreads();   // previous iteration done reading vt
    {  // cooperative transpose-stage of V rows k0..k0+31 into vt[d][key]
      const int r = t >> 2;                 // key row 0..31
      const int cg = (t & 3) * 16;          // d group
      const uint4* src =
          reinterpret_cast<const uint4*>(vb + (baseRow + k0 + r) * D_MODEL + dh + cg);
      unsigned short tmp[16];
      *reinterpret_cast<uint4*>(tmp)     = src[0];
      *reinterpret_cast<uint4*>(tmp + 8) = src[1];
      if (kt2 < 31)   // prefetch next V tile slice
        __builtin_prefetch(src + (32 * D_MODEL) / 8, 0, 1);
#pragma unroll
      for (int j = 0; j < 16; ++j) vt[(cg + j) * LSTR + r] = tmp[j];
    }
    __syncthreads();

#pragma unroll
    for (int sub = 0; sub < 2; ++sub) {
      const int n0 = k0 + sub * 16;
      Frag16 bk[2];
      const unsigned short* krow = kb + (baseRow + n0 + laneLo) * D_MODEL + dh;
#pragma unroll
      for (int c = 0; c < 2; ++c) {
        const uint4* p = reinterpret_cast<const uint4*>(krow + c * 32 + laneHi * 16);
        bk[c].q[0] = p[0];
        bk[c].q[1] = p[1];
      }
      v8f l = v8f_zero();
      l = wmma_bf16(aq[0].v, bk[0].v, l);
      l = wmma_bf16(aq[1].v, bk[1].v, l);
#pragma unroll
      for (int i = 0; i < 8; ++i) {
        const float x = l[i] * 0.125f;
        const float pv = __expf(x - rmax[i]) * rinv[i];
        const int qr = q0 + laneHi * 8 + i;
        scores[(((size_t)(bb * NH + h)) * SEQ + qr) * SEQ + (n0 + laneLo)] = pv;
        myPt[(laneHi * 8 + i) * LSTR + sub * 16 + laneLo] = f32_to_bf16(pv);
      }
    }

    // same-wave LDS store->load: DS ops are in-order; be explicit anyway
    asm volatile("s_wait_dscnt 0" ::: "memory");

    // P A-fragment (16x32) from per-wave LDS tile
    Frag16 ap;
    const uint4* pp = reinterpret_cast<const uint4*>(myPt + laneLo * LSTR + laneHi * 16);
    ap.q[0] = pp[0];
    ap.q[1] = pp[1];

#pragma unroll
    for (int nt = 0; nt < 4; ++nt) {   // V B-fragments from transposed tile
      Frag16 bv;
      const uint4* vp =
          reinterpret_cast<const uint4*>(vt + (nt * 16 + laneLo) * LSTR + laneHi * 16);
      bv.q[0] = vp[0];
      bv.q[1] = vp[1];
      oacc[nt] = wmma_bf16(ap.v, bv.v, oacc[nt]);
    }
  }

  // store attn as bf16 in [b*S + s, h*64 + d] layout for the output projection
#pragma unroll
  for (int nt = 0; nt < 4; ++nt) {
#pragma unroll
    for (int i = 0; i < 8; ++i) {
      const int qr = q0 + laneHi * 8 + i;
      attnb[(baseRow + qr) * D_MODEL + dh + nt * 16 + laneLo] = f32_to_bf16(oacc[nt][i]);
    }
  }
}

// ---------------------------------------------------------------------------
extern "C" void kernel_launch(void* const* d_in, const int* in_sizes, int n_in,
                              void* d_out, int out_size, void* d_ws, size_t ws_size,
                              hipStream_t stream) {
  (void)in_sizes; (void)n_in; (void)out_size; (void)ws_size;
  const float* query = (const float*)d_in[0];
  const float* key_  = (const float*)d_in[1];
  const float* value = (const float*)d_in[2];
  const float* Wq = (const float*)d_in[3];  const float* bq = (const float*)d_in[4];
  const float* Wk = (const float*)d_in[5];  const float* bk = (const float*)d_in[6];
  const float* Wv = (const float*)d_in[7];  const float* bv = (const float*)d_in[8];
  const float* Wo = (const float*)d_in[9];  const float* bo = (const float*)d_in[10];

  float* out    = (float*)d_out;                         // [8,1024,1024]
  float* scores = out + (size_t)NB * SEQ * D_MODEL;      // [8,16,1024,1024]

  const size_t ACT = (size_t)NB * SEQ * D_MODEL;         // 8.4M elements
  const size_t WEL = (size_t)D_MODEL * D_MODEL;          // 1M elements

  unsigned short* p = (unsigned short*)d_ws;
  unsigned short* xq = p;            p += ACT;   // converted activations
  unsigned short* xk = p;            p += ACT;
  unsigned short* xv = p;            p += ACT;
  unsigned short* wq = p;            p += WEL;   // converted weights
  unsigned short* wk = p;            p += WEL;
  unsigned short* wv = p;            p += WEL;
  unsigned short* wo = p;            p += WEL;
  unsigned short* qb = p;            p += ACT;   // projected q/k/v (bf16)
  unsigned short* kb = p;            p += ACT;
  unsigned short* vb = p;            p += ACT;
  unsigned short* attnb = p;         p += ACT;

  const dim3 cblk(256), cgrd(1024);
  cvt_f32_bf16<<<cgrd, cblk, 0, stream>>>(query, xq, (int)(ACT / 4));
  cvt_f32_bf16<<<cgrd, cblk, 0, stream>>>(key_,  xk, (int)(ACT / 4));
  cvt_f32_bf16<<<cgrd, cblk, 0, stream>>>(value, xv, (int)(ACT / 4));
  cvt_f32_bf16<<<cgrd, cblk, 0, stream>>>(Wq, wq, (int)(WEL / 4));
  cvt_f32_bf16<<<cgrd, cblk, 0, stream>>>(Wk, wk, (int)(WEL / 4));
  cvt_f32_bf16<<<cgrd, cblk, 0, stream>>>(Wv, wv, (int)(WEL / 4));
  cvt_f32_bf16<<<cgrd, cblk, 0, stream>>>(Wo, wo, (int)(WEL / 4));

  const dim3 blk(128);
  const dim3 gProj(NB * SEQ / 64, D_MODEL / 64);   // 128 x 16
  gemm_bf16<false><<<gProj, blk, 0, stream>>>(xq, wq, bq, qb);
  gemm_bf16<false><<<gProj, blk, 0, stream>>>(xk, wk, bk, kb);
  gemm_bf16<false><<<gProj, blk, 0, stream>>>(xv, wv, bv, vb);

  const dim3 gAttn(SEQ / 64, NH, NB);              // 16 x 16 x 8
  attention<<<gAttn, blk, 0, stream>>>(qb, kb, vb, scores, attnb);

  gemm_bf16<true><<<gProj, blk, 0, stream>>>(attnb, wo, bo, out);
}
